// SiamFCDemo_12429635354869
// MI455X (gfx1250) — matrix operand order repaired and checked by reading, not compiled, not run on hardware
//
#include <hip/hip_runtime.h>
#include <hip/hip_bf16.h>

// CDNA5 / gfx1250 SiamFC pipeline. All conv layers + xcorr via
// v_wmma_f32_16x16x32_f16 (f16 operands, f32 accumulate).
// Activations are NHWC f16 (C=32 == one WMMA K-slice).

typedef __attribute__((ext_vector_type(16))) _Float16 v16h;
typedef __attribute__((ext_vector_type(8)))  _Float16 v8h;
typedef __attribute__((ext_vector_type(8)))  float    v8f;

#define WMMA16(a, b, c) \
  __builtin_amdgcn_wmma_f32_16x16x32_f16(false, (a), false, (b), (short)0, (c), false, false)

__device__ __forceinline__ int imin(int a, int b) { return a < b ? a : b; }

// ---------------------------------------------------------------------------
// BN fold: y = conv*s + t, s = g*rsqrt(v+eps), t = (bias-mean)*s + beta
// ---------------------------------------------------------------------------
__global__ void sfc_prep_bn(const float* __restrict__ cb, const float* __restrict__ g,
                            const float* __restrict__ bb, const float* __restrict__ m,
                            const float* __restrict__ v, float* __restrict__ s,
                            float* __restrict__ t) {
  int c = threadIdx.x;  // 32 threads
  float inv = rsqrtf(v[c] + 1e-5f);
  float sc = g[c] * inv;
  s[c] = sc;
  t[c] = (cb[c] - m[c]) * sc + bb[c];
}

// ---------------------------------------------------------------------------
// Repack OIHW f32 weights (I=32) into WMMA A-operand layout, per tap:
//   wp[((tap*2+mt)*32+lane)*16 + j] = w[m = mt*16+(lane&15)][K_A(lane,j)][ky][kx]
//   K_A(lane,j) = j + (lane>=16 ? 16 : 0)
// ---------------------------------------------------------------------------
__global__ void sfc_prep_wc(const float* __restrict__ w, _Float16* __restrict__ wp, int KS) {
  int tid = blockIdx.x * blockDim.x + threadIdx.x;
  if (tid >= KS * KS * 64) return;
  int lane = tid & 31;
  int mt   = (tid >> 5) & 1;
  int tap  = tid >> 6;
  int ky = tap / KS, kx = tap - ky * KS;
  int mrow = mt * 16 + (lane & 15);
  int hi = (lane >> 4) & 1;
#pragma unroll
  for (int j = 0; j < 16; ++j) {
    int k = j + hi * 16;
    wp[(size_t)tid * 16 + j] = (_Float16)w[((mrow * 32 + k) * KS + ky) * KS + kx];
  }
}

// conv1 weights: K = c*121 + ky*11 + kx  (== flat OIHW order for I=3), padded 363->384
__global__ void sfc_prep_w1(const float* __restrict__ w, _Float16* __restrict__ wp) {
  int tid = blockIdx.x * blockDim.x + threadIdx.x;
  if (tid >= 12 * 64) return;
  int lane = tid & 31;
  int chunk = tid >> 6;
  int mt = (tid >> 5) & 1;
  int mrow = mt * 16 + (lane & 15);
  int hi = (lane >> 4) & 1;
#pragma unroll
  for (int j = 0; j < 16; ++j) {
    int k = chunk * 32 + j + hi * 16;
    wp[(size_t)tid * 16 + j] = (k < 363) ? (_Float16)w[mrow * 363 + k] : (_Float16)0.f;
  }
}

// ---------------------------------------------------------------------------
// Fused BN+ReLU store of two 16x16 f32 accumulators (out-ch 0..15 / 16..31)
// into NHWC f16.
// ---------------------------------------------------------------------------
__device__ __forceinline__ void sfc_store_nhwc(_Float16* __restrict__ out,
                                               const float* __restrict__ s,
                                               const float* __restrict__ t,
                                               long rowbase, int ox, int Wo, int hi,
                                               v8f a0, v8f a1) {
  if (ox < Wo) {
    int mb = hi * 8;
    v8h o0, o1;
#pragma unroll
    for (int r = 0; r < 8; ++r) {
      int c0 = mb + r;
      float y0 = a0[r] * s[c0] + t[c0];
      o0[r] = (_Float16)(y0 > 0.f ? y0 : 0.f);
      int c1 = 16 + mb + r;
      float y1 = a1[r] * s[c1] + t[c1];
      o1[r] = (_Float16)(y1 > 0.f ? y1 : 0.f);
    }
    _Float16* op = out + (rowbase + ox) * 32;
    *(v8h*)(op + mb) = o0;
    *(v8h*)(op + 16 + mb) = o1;
  }
}

// ---------------------------------------------------------------------------
// conv1: 11x11 stride-2, C_in=3, NCHW f32 in -> NHWC f16 out, BN+ReLU fused.
// LDS-staged patch; K padded 363->12x32. B-build offsets are compile-time
// immediates (lane-half branch removes all runtime k->(c,ky,kx) arithmetic).
// ---------------------------------------------------------------------------
#define C1_LW 41

template <int K>
__device__ __forceinline__ _Float16 c1_fetch(const _Float16* sm, int twoN) {
  if constexpr (K >= 363) {
    return (_Float16)0.f;
  } else {
    constexpr int c = K / 121;
    constexpr int rr = K % 121;
    constexpr int ky = rr / 11;
    constexpr int kx = rr % 11;
    constexpr int off = (c * 11 + ky) * C1_LW + kx;
    return sm[off + twoN];
  }
}

template <int CHUNK, int SEL>
__device__ __forceinline__ v16h c1_build(const _Float16* sm, int twoN) {
  v16h b;
  b[0]  = c1_fetch<CHUNK * 32 + 0  + SEL>(sm, twoN);
  b[1]  = c1_fetch<CHUNK * 32 + 1  + SEL>(sm, twoN);
  b[2]  = c1_fetch<CHUNK * 32 + 2  + SEL>(sm, twoN);
  b[3]  = c1_fetch<CHUNK * 32 + 3  + SEL>(sm, twoN);
  b[4]  = c1_fetch<CHUNK * 32 + 4  + SEL>(sm, twoN);
  b[5]  = c1_fetch<CHUNK * 32 + 5  + SEL>(sm, twoN);
  b[6]  = c1_fetch<CHUNK * 32 + 6  + SEL>(sm, twoN);
  b[7]  = c1_fetch<CHUNK * 32 + 7  + SEL>(sm, twoN);
  b[8]  = c1_fetch<CHUNK * 32 + 8  + SEL>(sm, twoN);
  b[9]  = c1_fetch<CHUNK * 32 + 9  + SEL>(sm, twoN);
  b[10] = c1_fetch<CHUNK * 32 + 10 + SEL>(sm, twoN);
  b[11] = c1_fetch<CHUNK * 32 + 11 + SEL>(sm, twoN);
  b[12] = c1_fetch<CHUNK * 32 + 12 + SEL>(sm, twoN);
  b[13] = c1_fetch<CHUNK * 32 + 13 + SEL>(sm, twoN);
  b[14] = c1_fetch<CHUNK * 32 + 14 + SEL>(sm, twoN);
  b[15] = c1_fetch<CHUNK * 32 + 15 + SEL>(sm, twoN);
  return b;
}

__global__ __launch_bounds__(32) void sfc_conv1_wmma(
    const float* __restrict__ in, const _Float16* __restrict__ wp,
    const float* __restrict__ s, const float* __restrict__ t,
    _Float16* __restrict__ out, int Hi, int Wi, int Ho, int Wo) {
  __shared__ _Float16 sm[3 * 11 * C1_LW];
  int lane = threadIdx.x;
  int n = blockIdx.z, oy = blockIdx.y, ox0 = blockIdx.x * 16;
  int iy0 = 2 * oy, ix0 = 2 * ox0;
  for (int idx = lane; idx < 3 * 11 * C1_LW; idx += 32) {
    int c = idx / (11 * C1_LW);
    int r2 = idx - c * 11 * C1_LW;
    int ry = r2 / C1_LW;
    int u = r2 - ry * C1_LW;
    int ix = imin(ix0 + u, Wi - 1);
    sm[idx] = (_Float16)in[((long)(n * 3 + c) * Hi + (iy0 + ry)) * Wi + ix];
  }
  __syncthreads();

  int Nn = lane & 15, hi = lane >> 4;
  int twoN = 2 * Nn;
  const _Float16* wl = wp + lane * 16;
  v8f acc0 = {}, acc1 = {};

#define C1_STEP(CH)                                                          \
  {                                                                          \
    v16h b = (lane < 16) ? c1_build<CH, 0>(sm, twoN)                         \
                         : c1_build<CH, 16>(sm, twoN);                       \
    v16h a0 = *(const v16h*)(wl + (CH * 64 + 0) * 16);                       \
    v16h a1 = *(const v16h*)(wl + (CH * 64 + 32) * 16);                      \
    acc0 = WMMA16(a0, b, acc0);                                              \
    acc1 = WMMA16(a1, b, acc1);                                              \
  }
  C1_STEP(0) C1_STEP(1) C1_STEP(2) C1_STEP(3) C1_STEP(4) C1_STEP(5)
  C1_STEP(6) C1_STEP(7) C1_STEP(8) C1_STEP(9) C1_STEP(10) C1_STEP(11)
#undef C1_STEP

  sfc_store_nhwc(out, s, t, (long)(n * Ho + oy) * Wo, ox0 + Nn, Wo, hi, acc0, acc1);
}

// ---------------------------------------------------------------------------
// 3x3 s2 maxpool on NHWC f16; packed f16 max (v_pk_max) over 8 channels/thread.
// ---------------------------------------------------------------------------
__global__ void sfc_maxpool(const _Float16* __restrict__ in, _Float16* __restrict__ out,
                            int Hi, int Wi, int Ho, int Wo, int total) {
  int idx = blockIdx.x * blockDim.x + threadIdx.x;
  if (idx >= total) return;
  int c8 = (idx & 3) * 8;
  int p = idx >> 2;
  int ox = p % Wo; p /= Wo;
  int oy = p % Ho;
  int n = p / Ho;
  int iy = 2 * oy, ix = 2 * ox;
  const _Float16* base = in + ((long)(n * Hi + iy) * Wi + ix) * 32 + c8;
  v8h mx = *(const v8h*)base;
#pragma unroll
  for (int dy = 0; dy < 3; ++dy)
#pragma unroll
    for (int dx = 0; dx < 3; ++dx) {
      if (dy == 0 && dx == 0) continue;
      v8h v = *(const v8h*)(base + ((long)dy * Wi + dx) * 32);
      mx = __builtin_elementwise_max(mx, v);
    }
  *(v8h*)(out + ((long)(n * Ho + oy) * Wo + ox) * 32 + c8) = mx;
}

// ---------------------------------------------------------------------------
// Generic KSxKS conv, C_in=C_out=32, implicit GEMM. Each wave computes a
// 32-column x 32-channel tile: two B tiles share one A pair -> 4 WMMAs/tap.
// ---------------------------------------------------------------------------
template <int KS>
__global__ __launch_bounds__(32) void sfc_conv_wmma(
    const _Float16* __restrict__ in, const _Float16* __restrict__ wp,
    const float* __restrict__ s, const float* __restrict__ t,
    _Float16* __restrict__ out, int Hi, int Wi, int Ho, int Wo) {
  int lane = threadIdx.x;
  int n = blockIdx.z, oy = blockIdx.y, ox0 = blockIdx.x * 32;
  int Nn = lane & 15, hi = lane >> 4;
  int oxA = imin(ox0 + Nn, Wo - 1);
  int oxB = imin(ox0 + 16 + Nn, Wo - 1);
  const _Float16* wl = wp + lane * 16;
  const _Float16* inb = in + (long)n * Hi * Wi * 32 + hi * 16;
  v8f accA0 = {}, accA1 = {}, accB0 = {}, accB1 = {};
#pragma unroll
  for (int ky = 0; ky < KS; ++ky) {
    const _Float16* row = inb + (long)(oy + ky) * Wi * 32;
#pragma unroll
    for (int kx = 0; kx < KS; ++kx) {
      int tap = ky * KS + kx;
      v16h bA = *(const v16h*)(row + (oxA + kx) * 32);
      v16h bB = *(const v16h*)(row + (oxB + kx) * 32);
      v16h a0 = *(const v16h*)(wl + (tap * 64 + 0) * 16);
      v16h a1 = *(const v16h*)(wl + (tap * 64 + 32) * 16);
      accA0 = WMMA16(a0, bA, accA0);
      accA1 = WMMA16(a1, bA, accA1);
      accB0 = WMMA16(a0, bB, accB0);
      accB1 = WMMA16(a1, bB, accB1);
    }
  }
  long rowbase = (long)(n * Ho + oy) * Wo;
  sfc_store_nhwc(out, s, t, rowbase, ox0 + Nn, Wo, hi, accA0, accA1);
  sfc_store_nhwc(out, s, t, rowbase, ox0 + 16 + Nn, Wo, hi, accB0, accB1);
}

// ---------------------------------------------------------------------------
// Channel-summed xcorr as a true GEMM over x rows:
//   For fixed x row iy:  D[ky][ox] = sum_{c,kx} z[c,ky,kx] * x[c,iy,ox+kx]
// is a real GEMM (A = z rows [M=ky 23->2 tiles], B = x row [K=c per kx chunk,
// N=16 ox]); D row ky belongs to output row oy = iy - ky. Scatter-accumulate
// D into a per-wave LDS ring (ds_add_f32, branch-free trash row for invalid
// ky/oy) and retire row iy-22 each step. 72% M-utilization vs 1/16 for the
// replication trick (~8-10x fewer WMMAs). Work split: 11 ox-tiles x 4 row
// chunks per sample (22-row iy halo recompute) -> 1408 waves.
// z staged into LDS via global_load_async_to_lds_b128 (ASYNCcnt path).
// ---------------------------------------------------------------------------
__global__ __launch_bounds__(256) void sfc_xcorr_wmma(const _Float16* __restrict__ z,
                                                      const _Float16* __restrict__ x,
                                                      float* __restrict__ out) {
  __shared__ _Float16 zs[23 * 23 * 32];     // 33856 B, NHWC z for sample n
  __shared__ float ring[8][33 * 16];        // per-wave ring: 32 rows + trash row

  int tid = threadIdx.x;
  int wave = tid >> 5, lane = tid & 31;
  int n = blockIdx.y;
  const _Float16* zn = z + (long)n * 23 * 23 * 32;

  // Async-stage z into LDS (2116 x b128), then drain ASYNCcnt.
  for (int i = tid; i < (23 * 23 * 32) / 8; i += 256) {
    unsigned ldsoff = (unsigned)(unsigned long long)(zs + i * 8);
    const void* g = (const void*)(zn + i * 8);
    asm volatile("global_load_async_to_lds_b128 %0, %1, off"
                 :: "v"(ldsoff), "v"(g) : "memory");
  }
  asm volatile("s_wait_asynccnt 0x0" ::: "memory");

  float* myring = ring[wave];
  for (int i = lane; i < 33 * 16; i += 32) myring[i] = 0.f;
  __syncthreads();

  int pair = blockIdx.x * 8 + wave;         // 11 ox-tiles x 4 row-chunks = 44
  if (pair >= 44) return;
  int tileox = pair % 11;
  int chunk = pair / 11;
  int ox0 = tileox * 16;
  int oyS = chunk * 41;
  int oyE = imin(oyS + 40, 160);

  int Nn = lane & 15, hi = lane >> 4;
  int m = Nn;                               // A-operand M row == lane&15
  int m1 = imin(16 + m, 22);                // tile1 rows (23..31 unused)
  int oxc = imin(ox0 + Nn, 160);
  const _Float16* xb = x + (long)n * 183 * 183 * 32 + hi * 8;
  long outb = (long)n * 161 * 161;

  for (int iy = oyS; iy <= oyE + 22; ++iy) {
    const _Float16* xr = xb + (long)iy * 183 * 32;
    __builtin_prefetch(xr + 183 * 32, 0, 1);     // next x row
    // Defeat LICM of the 46 loop-invariant A vectors (would spill).
    const _Float16* zsp = zs;
    asm volatile("" : "+v"(zsp));

    v8f acc0 = {}, acc1 = {};
#pragma unroll
    for (int kx = 0; kx < 23; ++kx) {
      int zo0 = (m * 23 + kx) * 32 + 8 * hi;     // A K layout: c in
      int zo1 = (m1 * 23 + kx) * 32 + 8 * hi;    // [8hi,8hi+8) u [16+8hi,24+8hi)
      v16h a0, a1;
      *(v8h*)&a0 = *(const v8h*)(zsp + zo0);
      *((v8h*)&a0 + 1) = *(const v8h*)(zsp + zo0 + 16);
      *(v8h*)&a1 = *(const v8h*)(zsp + zo1);
      *((v8h*)&a1 + 1) = *(const v8h*)(zsp + zo1 + 16);
      v16h b;
      *(v8h*)&b = *(const v8h*)(xr + (oxc + kx) * 32);
      *((v8h*)&b + 1) = *(const v8h*)(xr + (oxc + kx) * 32 + 16);
      acc0 = WMMA16(a0, b, acc0);
      acc1 = WMMA16(a1, b, acc1);
    }

    // Scatter D rows into the ring (oy = iy - ky); invalid -> trash row 32.
#pragma unroll
    for (int r = 0; r < 8; ++r) {
      int ky = 8 * hi + r;                       // tile0: ky 0..15, all valid
      int oy = iy - ky;
      bool ok = (oy >= oyS) && (oy <= oyE);
      int slot = ok ? (oy & 31) : 32;
      atomicAdd(&myring[slot * 16 + Nn], acc0[r]);
    }
#pragma unroll
    for (int r = 0; r < 7; ++r) {
      int ky = 16 + 8 * hi + r;                  // tile1: ky 16..22 (hi=0 only)
      int oy = iy - ky;
      bool ok = (hi == 0) && (oy >= oyS) && (oy <= oyE);
      int slot = ok ? (oy & 31) : 32;
      atomicAdd(&myring[slot * 16 + Nn], acc1[r]);
    }

    // Retire completed output row oy = iy - 22.
    int oyd = iy - 22;
    if (oyd >= oyS) {
      int slot = oyd & 31;
      if (lane < 16) {
        float vsum = myring[slot * 16 + lane];
        myring[slot * 16 + lane] = 0.f;
        int ox = ox0 + lane;
        if (ox < 161) out[outb + (long)oyd * 161 + ox] = vsum;
      }
    }
  }
}

// ---------------------------------------------------------------------------
// Host launch
// ---------------------------------------------------------------------------
extern "C" void kernel_launch(void* const* d_in, const int* in_sizes, int n_in,
                              void* d_out, int out_size, void* d_ws, size_t ws_size,
                              hipStream_t stream) {
  (void)in_sizes; (void)n_in; (void)out_size; (void)ws_size;
  const float* z_img = (const float*)d_in[0];
  const float* x_img = (const float*)d_in[1];
  const float* c1w = (const float*)d_in[2];  const float* c1b = (const float*)d_in[3];
  const float* g1  = (const float*)d_in[4];  const float* b1  = (const float*)d_in[5];
  const float* m1  = (const float*)d_in[6];  const float* v1  = (const float*)d_in[7];
  const float* c2w = (const float*)d_in[8];  const float* c2b = (const float*)d_in[9];
  const float* g2  = (const float*)d_in[10]; const float* b2  = (const float*)d_in[11];
  const float* m2  = (const float*)d_in[12]; const float* v2  = (const float*)d_in[13];
  const float* c3w = (const float*)d_in[14]; const float* c3b = (const float*)d_in[15];
  const float* g3  = (const float*)d_in[16]; const float* b3  = (const float*)d_in[17];
  const float* m3  = (const float*)d_in[18]; const float* v3  = (const float*)d_in[19];
  float* out = (float*)d_out;

  // Workspace layout (~439 MB): weights + bn + activation buffers (reused).
  char* ws = (char*)d_ws;
  size_t off = 0;
  auto take = [&](size_t b) { size_t r = off; off = (off + b + 255) & ~(size_t)255; return r; };
  _Float16* w1p  = (_Float16*)(ws + take(12 * 64 * 16 * 2));   // 24576 B
  _Float16* w2p  = (_Float16*)(ws + take(25 * 64 * 16 * 2));   // 51200 B
  _Float16* w3p  = (_Float16*)(ws + take(9 * 64 * 16 * 2));    // 18432 B
  float*    bn   = (float*)(ws + take(6 * 32 * 4));            // s1,t1,s2,t2,s3,t3
  _Float16* bufA = (_Float16*)(ws + take(294176768ull));       // conv1 out (x: 379^2)
  _Float16* bufB = (_Float16*)(ws + take(73156608ull));        // pool out / conv3-x out
  _Float16* bufC = (_Float16*)(ws + take(70092800ull));        // conv2 out
  _Float16* bufE = (_Float16*)(ws + take(1083392ull));         // z conv3 out (23^2)

  float *s1 = bn, *t1 = bn + 32, *s2 = bn + 64, *t2 = bn + 96, *s3 = bn + 128, *t3 = bn + 160;
  sfc_prep_bn<<<1, 32, 0, stream>>>(c1b, g1, b1, m1, v1, s1, t1);
  sfc_prep_bn<<<1, 32, 0, stream>>>(c2b, g2, b2, m2, v2, s2, t2);
  sfc_prep_bn<<<1, 32, 0, stream>>>(c3b, g3, b3, m3, v3, s3, t3);
  sfc_prep_w1<<<(12 * 64 + 63) / 64, 64, 0, stream>>>(c1w, w1p);
  sfc_prep_wc<<<(25 * 64 + 63) / 64, 64, 0, stream>>>(c2w, w2p, 5);
  sfc_prep_wc<<<(9 * 64 + 63) / 64, 64, 0, stream>>>(c3w, w3p, 3);

  // ---- z branch: 127 -> 59 -> 29 -> 25 -> 23 ----
  sfc_conv1_wmma<<<dim3(4, 59, 32), 32, 0, stream>>>(z_img, w1p, s1, t1, bufA, 127, 127, 59, 59);
  { int tot = 32 * 29 * 29 * 4;
    sfc_maxpool<<<(tot + 255) / 256, 256, 0, stream>>>(bufA, bufB, 59, 59, 29, 29, tot); }
  sfc_conv_wmma<5><<<dim3(1, 25, 32), 32, 0, stream>>>(bufB, w2p, s2, t2, bufC, 29, 29, 25, 25);
  sfc_conv_wmma<3><<<dim3(1, 23, 32), 32, 0, stream>>>(bufC, w3p, s3, t3, bufE, 25, 25, 23, 23);

  // ---- x branch: 767 -> 379 -> 189 -> 185 -> 183 ----
  sfc_conv1_wmma<<<dim3(24, 379, 32), 32, 0, stream>>>(x_img, w1p, s1, t1, bufA, 767, 767, 379, 379);
  { int tot = 32 * 189 * 189 * 4;
    sfc_maxpool<<<(tot + 255) / 256, 256, 0, stream>>>(bufA, bufB, 379, 379, 189, 189, tot); }
  sfc_conv_wmma<5><<<dim3(6, 185, 32), 32, 0, stream>>>(bufB, w2p, s2, t2, bufC, 189, 189, 185, 185);
  sfc_conv_wmma<3><<<dim3(6, 183, 32), 32, 0, stream>>>(bufC, w3p, s3, t3, bufB, 185, 185, 183, 183);

  // ---- xcorr: 44 wave-jobs per sample (11 ox-tiles x 4 row chunks) ----
  sfc_xcorr_wmma<<<dim3(6, 32), 256, 0, stream>>>(bufE, bufB, out);
}
